// DeformableTinyImageNet_55645596287303
// MI455X (gfx1250) — matrix operand order repaired and checked by reading, not compiled, not run on hardware
//
#include <hip/hip_runtime.h>
#include <hip/hip_bf16.h>

typedef __attribute__((ext_vector_type(16))) _Float16 v16h;
typedef __attribute__((ext_vector_type(8)))  float    v8f;

#define EPSV 1e-5f

__device__ __forceinline__ int iclamp(int v, int lo, int hi) {
    return v < lo ? lo : (v > hi ? hi : v);
}

__device__ __forceinline__ v16h splat16(_Float16 s) {
    v16h r;
#pragma unroll
    for (int i = 0; i < 16; ++i) r[i] = s;
    return r;
}

// ---------------------------------------------------------------------------
// Kernel P: pack conv2_w (128,64,3,3) fp32 -> WMMA B-fragment fp16 layout.
// K order: k = kk*64 + c  (kk = ki*3+kj). Fragment layout:
//   bpack[ ((nt*18 + kc)*32 + lane)*16 + hh ],  k = kc*32 + (lane>>4)*16 + hh,
//   n = nt*16 + (lane&15).
// Also zeroes the pooled accumulator (32*128 floats).
// ---------------------------------------------------------------------------
__global__ __launch_bounds__(256) void prep_kernel(const float* __restrict__ w2,
                                                   _Float16* __restrict__ bpack,
                                                   float* __restrict__ pooled) {
    int tid = blockIdx.x * 256 + threadIdx.x;
    if (tid < 8 * 18 * 32 * 16) {
        int hh   = tid & 15;
        int lane = (tid >> 4) & 31;
        int r    = tid >> 9;
        int kc   = r % 18;
        int nt   = r / 18;
        int n    = nt * 16 + (lane & 15);
        int gg   = lane >> 4;
        int k    = kc * 32 + gg * 16 + hh;
        int kk   = k >> 6;   // /64
        int c    = k & 63;
        bpack[tid] = (_Float16)w2[((size_t)(n * 64 + c)) * 9 + kk];
    }
    if (tid < 32 * 128) pooled[tid] = 0.f;
}

// ---------------------------------------------------------------------------
// Kernel A: offset1 conv  (B,3,64,64) -> (B,18,64,64), 3x3 SAME, fp32.
// ---------------------------------------------------------------------------
__global__ __launch_bounds__(256) void offset1_conv(const float* __restrict__ x,
                                                    const float* __restrict__ w,
                                                    const float* __restrict__ bias,
                                                    float* __restrict__ out) {
    __shared__ float Wl[27 * 18];
    for (int i = threadIdx.x; i < 27 * 18; i += 256) {
        int o = i % 18, t = i / 18, c = t / 9, kk = t % 9;
        Wl[i] = w[((size_t)(o * 3 + c)) * 9 + kk];
    }
    __syncthreads();
    int tid = blockIdx.x * 256 + threadIdx.x;
    int b = tid >> 12, hw = tid & 4095, h = hw >> 6, wc = hw & 63;
    float acc[18];
#pragma unroll
    for (int o = 0; o < 18; ++o) acc[o] = bias[o];
    for (int kk = 0; kk < 9; ++kk) {
        int ki = kk / 3, kj = kk % 3;
        int y = h + ki - 1, xx = wc + kj - 1;
        if ((unsigned)y < 64u && (unsigned)xx < 64u) {
#pragma unroll
            for (int c = 0; c < 3; ++c) {
                float v = x[((size_t)(b * 3 + c)) * 4096 + y * 64 + xx];
                const float* wl = &Wl[(c * 9 + kk) * 18];
#pragma unroll
                for (int o = 0; o < 18; ++o) acc[o] += v * wl[o];
            }
        }
    }
#pragma unroll
    for (int o = 0; o < 18; ++o) out[((size_t)(b * 18 + o)) * 4096 + hw] = acc[o];
}

// ---------------------------------------------------------------------------
// Kernel B: deform conv1 (3->64) + BN1 + ReLU, writes h1 in NHWC fp16.
// ---------------------------------------------------------------------------
__global__ __launch_bounds__(256) void deform1_kernel(const float* __restrict__ x,
                                                      const float* __restrict__ off1,
                                                      const float* __restrict__ w,
                                                      const float* __restrict__ bias,
                                                      const float* __restrict__ gamma,
                                                      const float* __restrict__ beta,
                                                      const float* __restrict__ mean,
                                                      const float* __restrict__ var,
                                                      _Float16* __restrict__ h1) {
    __shared__ float Wl[27 * 64];
    __shared__ float scl[64], shf[64];
    for (int i = threadIdx.x; i < 27 * 64; i += 256) {
        int co = i & 63, t = i >> 6, c = t / 9, kk = t % 9;
        Wl[i] = w[((size_t)(co * 3 + c)) * 9 + kk];
    }
    if (threadIdx.x < 64) {
        int co = threadIdx.x;
        float s = gamma[co] * rsqrtf(var[co] + EPSV);
        scl[co] = s;
        shf[co] = (bias[co] - mean[co]) * s + beta[co];
    }
    __syncthreads();
    int tid = blockIdx.x * 256 + threadIdx.x;
    int b = tid >> 12, hw = tid & 4095, h = hw >> 6, wc = hw & 63;
    float acc[64];
#pragma unroll
    for (int co = 0; co < 64; ++co) acc[co] = 0.f;

    for (int kk = 0; kk < 9; ++kk) {
        float dy = off1[((size_t)(b * 18 + 2 * kk)) * 4096 + hw];
        float dx = off1[((size_t)(b * 18 + 2 * kk + 1)) * 4096 + hw];
        int ki = kk / 3, kj = kk % 3;
        float py = (float)h + (float)(ki - 1) + dy;
        float px = (float)wc + (float)(kj - 1) + dx;
        float fy = floorf(py), fx = floorf(px);
        int y0 = (int)fy, x0 = (int)fx;
        float wy = py - fy, wx = px - fx;
        bool y0v = (unsigned)y0 < 64u, y1v = (unsigned)(y0 + 1) < 64u;
        bool x0v = (unsigned)x0 < 64u, x1v = (unsigned)(x0 + 1) < 64u;
        int y0c = iclamp(y0, 0, 63), y1c = iclamp(y0 + 1, 0, 63);
        int x0c = iclamp(x0, 0, 63), x1c = iclamp(x0 + 1, 0, 63);
        float w00 = (1.f - wy) * (1.f - wx) * ((y0v && x0v) ? 1.f : 0.f);
        float w01 = (1.f - wy) * wx * ((y0v && x1v) ? 1.f : 0.f);
        float w10 = wy * (1.f - wx) * ((y1v && x0v) ? 1.f : 0.f);
        float w11 = wy * wx * ((y1v && x1v) ? 1.f : 0.f);
#pragma unroll
        for (int c = 0; c < 3; ++c) {
            const float* xb = x + ((size_t)(b * 3 + c)) * 4096;
            float v = w00 * xb[y0c * 64 + x0c] + w01 * xb[y0c * 64 + x1c] +
                      w10 * xb[y1c * 64 + x0c] + w11 * xb[y1c * 64 + x1c];
            const float* wl = &Wl[(c * 9 + kk) * 64];
#pragma unroll
            for (int co = 0; co < 64; ++co) acc[co] += v * wl[co];
        }
    }
    _Float16* ob = h1 + (size_t)tid * 64;
#pragma unroll
    for (int co = 0; co < 64; ++co)
        ob[co] = (_Float16)fmaxf(acc[co] * scl[co] + shf[co], 0.f);
}

// ---------------------------------------------------------------------------
// Kernel C: offset2 conv (64->18), reads NHWC fp16 h1, LDS-staged weights.
// ---------------------------------------------------------------------------
__global__ __launch_bounds__(256) void offset2_conv(const _Float16* __restrict__ h1,
                                                    const float* __restrict__ w,
                                                    const float* __restrict__ bias,
                                                    float* __restrict__ out) {
    __shared__ float Wl[9 * 64 * 18];  // [kk][c][o]
    for (int i = threadIdx.x; i < 9 * 64 * 18; i += 256) {
        int o = i % 18, t = i / 18, c = t % 64, kk = t / 64;
        Wl[i] = w[((size_t)(o * 64 + c)) * 9 + kk];
    }
    __syncthreads();
    int tid = blockIdx.x * 256 + threadIdx.x;
    int b = tid >> 12, hw = tid & 4095, h = hw >> 6, wc = hw & 63;
    float acc[18];
#pragma unroll
    for (int o = 0; o < 18; ++o) acc[o] = bias[o];
    for (int kk = 0; kk < 9; ++kk) {
        int ki = kk / 3, kj = kk % 3;
        int y = h + ki - 1, xx = wc + kj - 1;
        if ((unsigned)y < 64u && (unsigned)xx < 64u) {
            const _Float16* p = h1 + ((size_t)(b * 4096 + y * 64 + xx)) * 64;
            const float* wl = &Wl[kk * 64 * 18];
            for (int c = 0; c < 64; ++c) {
                float v = (float)p[c];
#pragma unroll
                for (int o = 0; o < 18; ++o) acc[o] += v * wl[c * 18 + o];
            }
        }
    }
#pragma unroll
    for (int o = 0; o < 18; ++o) out[((size_t)(b * 18 + o)) * 4096 + hw] = acc[o];
}

// ---------------------------------------------------------------------------
// Kernel D: deform conv2 (64->128) as WMMA implicit GEMM, fused with
//           bias + BN2 + ReLU + global-average-pool (sum via atomics).
// M = 131072 spatial positions, N = 128, K = 576 (k = kk*64 + c).
// 8 waves/block; each wave: 16-position M-tile x 128 channels
// (8 x v8f accumulators), 2 K-chunks x 8 N-tiles = 16 WMMAs per 3x3 tap.
// Bilinear im2col build uses packed fp16 vector math (v_pk_*_f16) so the
// sampling VALU co-executes efficiently against the XDL WMMA pipe.
// ---------------------------------------------------------------------------
__global__ __launch_bounds__(256) void deform2_wmma(const _Float16* __restrict__ h1,
                                                    const float* __restrict__ off2,
                                                    const _Float16* __restrict__ bpack,
                                                    const float* __restrict__ bias,
                                                    const float* __restrict__ gamma,
                                                    const float* __restrict__ beta,
                                                    const float* __restrict__ mean,
                                                    const float* __restrict__ var,
                                                    float* __restrict__ pooled) {
    __shared__ _Float16 Asl[8 * 16 * 64];  // per-wave 16x64 fp16 im2col slice

    const int wave = threadIdx.x >> 5;
    const int lane = threadIdx.x & 31;
    const int sbase = blockIdx.x * 128 + wave * 16;
    const int b = sbase >> 12;  // whole tile in one image (4096 % 16 == 0)

    // sampling role: 2 lanes per position, 32 channels each
    const int pos = lane >> 1;
    const int cb  = (lane & 1) * 32;
    const int s   = sbase + pos;
    const int hw  = s & 4095;
    const int hrow = hw >> 6;
    const int wcol = hw & 63;

    // WMMA fragment role
    const int g = lane >> 4;
    const int M = lane & 15;
    const unsigned* a32 = (const unsigned*)Asl + ((wave * 1024 + M * 64) >> 1);
    const v16h* bvec = (const v16h*)bpack;
    const _Float16* hb = h1 + (((size_t)b) << 12) * 64;

    v8f acc[8];
#pragma unroll
    for (int i = 0; i < 8; ++i)
#pragma unroll
        for (int r = 0; r < 8; ++r) acc[i][r] = 0.f;

    for (int kk = 0; kk < 9; ++kk) {
        // ---- build 16x64 fp16 im2col slice for this tap (bilinear, NHWC) ----
        float dy = off2[((size_t)(b * 18 + 2 * kk)) * 4096 + hw];
        float dx = off2[((size_t)(b * 18 + 2 * kk + 1)) * 4096 + hw];
        int ki = kk / 3, kj = kk % 3;
        float py = (float)hrow + (float)(ki - 1) + dy;
        float px = (float)wcol + (float)(kj - 1) + dx;
        float fy = floorf(py), fx = floorf(px);
        int y0 = (int)fy, x0 = (int)fx;
        float wy = py - fy, wx = px - fx;
        bool y0v = (unsigned)y0 < 64u, y1v = (unsigned)(y0 + 1) < 64u;
        bool x0v = (unsigned)x0 < 64u, x1v = (unsigned)(x0 + 1) < 64u;
        int y0c = iclamp(y0, 0, 63), y1c = iclamp(y0 + 1, 0, 63);
        int x0c = iclamp(x0, 0, 63), x1c = iclamp(x0 + 1, 0, 63);
        float w00 = (1.f - wy) * (1.f - wx) * ((y0v && x0v) ? 1.f : 0.f);
        float w01 = (1.f - wy) * wx * ((y0v && x1v) ? 1.f : 0.f);
        float w10 = wy * (1.f - wx) * ((y1v && x0v) ? 1.f : 0.f);
        float w11 = wy * wx * ((y1v && x1v) ? 1.f : 0.f);

        const v16h* p00 = (const v16h*)(hb + ((size_t)(y0c * 64 + x0c)) * 64 + cb);
        const v16h* p01 = (const v16h*)(hb + ((size_t)(y0c * 64 + x1c)) * 64 + cb);
        const v16h* p10 = (const v16h*)(hb + ((size_t)(y1c * 64 + x0c)) * 64 + cb);
        const v16h* p11 = (const v16h*)(hb + ((size_t)(y1c * 64 + x1c)) * 64 + cb);
        v16h r00a = p00[0], r00b = p00[1];
        v16h r01a = p01[0], r01b = p01[1];
        v16h r10a = p10[0], r10b = p10[1];
        v16h r11a = p11[0], r11b = p11[1];

        // packed fp16 blend: lowers to v_pk_mul_f16 / v_pk_fma_f16
        v16h s00 = splat16((_Float16)w00);
        v16h s01 = splat16((_Float16)w01);
        v16h s10 = splat16((_Float16)w10);
        v16h s11 = splat16((_Float16)w11);
        v16h oa = r00a * s00 + r01a * s01 + r10a * s10 + r11a * s11;
        v16h ob = r00b * s00 + r01b * s01 + r10b * s10 + r11b * s11;

        *(v16h*)&Asl[wave * 1024 + pos * 64 + cb]      = oa;
        *(v16h*)&Asl[wave * 1024 + pos * 64 + cb + 16] = ob;
        __syncthreads();

        // prefetch next tap's first B fragments into L2
        if (kk < 8)
            __builtin_prefetch((const void*)(bvec + ((kk * 2 + 2) * 32 + lane)), 0, 2);

        // ---- two K=32 chunks of WMMA over all 8 N-tiles ----
#pragma unroll
        for (int half = 0; half < 2; ++half) {
            union { v16h h; unsigned u[8]; } af;
#pragma unroll
            for (int v = 0; v < 8; ++v) {
                // contiguous per-lane K pattern (k = 16*g + s), matches bpack
                int base = half * 32 + 16 * g + 2 * v;
                af.u[v] = a32[base >> 1];
            }
            int kc = kk * 2 + half;
#pragma unroll
            for (int nt = 0; nt < 8; ++nt) {
                v16h bf = bvec[(nt * 18 + kc) * 32 + lane];
                acc[nt] = __builtin_amdgcn_wmma_f32_16x16x32_f16(
                    false, af.h, false, bf, (short)0, acc[nt], false, false);
            }
        }
        __syncthreads();
    }

    // ---- epilogue: bias + BN + ReLU, reduce 8 rows in-lane, pool via atomics ----
    const int co0 = lane & 15;
#pragma unroll
    for (int nt = 0; nt < 8; ++nt) {
        int co = nt * 16 + co0;
        float scale = gamma[co] * rsqrtf(var[co] + EPSV);
        float shift = (bias[co] - mean[co]) * scale + beta[co];
        float sum = 0.f;
#pragma unroll
        for (int r = 0; r < 8; ++r)
            sum += fmaxf(acc[nt][r] * scale + shift, 0.f);
        atomicAdd(&pooled[b * 128 + co], sum);
    }
}

// ---------------------------------------------------------------------------
// Kernel E: pooled/4096 -> fc1 + ReLU -> fc2 -> out (32 x 200). One block.
// ---------------------------------------------------------------------------
__global__ __launch_bounds__(256) void head_kernel(const float* __restrict__ pooled,
                                                   const float* __restrict__ f1w,
                                                   const float* __restrict__ f1b,
                                                   const float* __restrict__ f2w,
                                                   const float* __restrict__ f2b,
                                                   float* __restrict__ out) {
    __shared__ float pool[32 * 128];
    __shared__ float hid[32 * 256];
    int t = threadIdx.x;
    for (int i = t; i < 32 * 128; i += 256) pool[i] = pooled[i] * (1.f / 4096.f);
    __syncthreads();
    {
        const float* wr = f1w + (size_t)t * 128;
        float bv = f1b[t];
        for (int bi = 0; bi < 32; ++bi) {
            float ssum = bv;
            for (int k = 0; k < 128; ++k) ssum += pool[bi * 128 + k] * wr[k];
            hid[bi * 256 + t] = fmaxf(ssum, 0.f);
        }
    }
    __syncthreads();
    if (t < 200) {
        const float* wr = f2w + (size_t)t * 256;
        float bv = f2b[t];
        for (int bi = 0; bi < 32; ++bi) {
            float ssum = bv;
            for (int k = 0; k < 256; ++k) ssum += hid[bi * 256 + k] * wr[k];
            out[bi * 200 + t] = ssum;
        }
    }
}

// ---------------------------------------------------------------------------
extern "C" void kernel_launch(void* const* d_in, const int* in_sizes, int n_in,
                              void* d_out, int out_size, void* d_ws, size_t ws_size,
                              hipStream_t stream) {
    const float* x   = (const float*)d_in[0];
    const float* o1w = (const float*)d_in[1];
    const float* o1b = (const float*)d_in[2];
    const float* c1w = (const float*)d_in[3];
    const float* c1b = (const float*)d_in[4];
    const float* g1  = (const float*)d_in[5];
    const float* be1 = (const float*)d_in[6];
    const float* m1  = (const float*)d_in[7];
    const float* v1  = (const float*)d_in[8];
    const float* o2w = (const float*)d_in[9];
    const float* o2b = (const float*)d_in[10];
    const float* c2w = (const float*)d_in[11];
    const float* c2b = (const float*)d_in[12];
    const float* g2  = (const float*)d_in[13];
    const float* be2 = (const float*)d_in[14];
    const float* m2  = (const float*)d_in[15];
    const float* v2  = (const float*)d_in[16];
    const float* f1w = (const float*)d_in[17];
    const float* f1b = (const float*)d_in[18];
    const float* f2w = (const float*)d_in[19];
    const float* f2b = (const float*)d_in[20];
    float* out = (float*)d_out;

    char* ws = (char*)d_ws;
    // workspace layout (all offsets 1 KiB aligned)
    float*    off1   = (float*)(ws + 0);                       //  9,437,184 B
    _Float16* h1     = (_Float16*)(ws + 9437184);              // 16,777,216 B
    float*    off2   = (float*)(ws + 26214400);                //  9,437,184 B
    _Float16* bpack  = (_Float16*)(ws + 35651584);             //    147,456 B
    float*    pooled = (float*)(ws + 35799040);                //     16,384 B

    prep_kernel<<<288, 256, 0, stream>>>(c2w, bpack, pooled);
    offset1_conv<<<512, 256, 0, stream>>>(x, o1w, o1b, off1);
    deform1_kernel<<<512, 256, 0, stream>>>(x, off1, c1w, c1b, g1, be1, m1, v1, h1);
    offset2_conv<<<512, 256, 0, stream>>>(h1, o2w, o2b, off2);
    deform2_wmma<<<1024, 256, 0, stream>>>(h1, off2, bpack, c2b, g2, be2, m2, v2, pooled);
    head_kernel<<<1, 256, 0, stream>>>(pooled, f1w, f1b, f2w, f2b, out);
}
